// SE_Block1D_7687991460027
// MI455X (gfx1250) — matrix operand order repaired and checked by reading, not compiled, not run on hardware
//
#include <hip/hip_runtime.h>
#include <stdint.h>

#define N_ROWS      16384
#define C_DIM       2048
#define H_DIM       128
#define K_TOP       32
#define ROWS_PER_WG 16
#define WAVES       8
#define THREADS     (WAVES * 32)

typedef __attribute__((ext_vector_type(16))) __bf16 v16bf;
typedef __attribute__((ext_vector_type(8)))  float  v8f;

union Frag32 {
    unsigned short us[16];
    uint4          q[2];
    v16bf          v;
};

__device__ __forceinline__ unsigned short f2bf(float f) {
    // round-to-nearest-even f32 -> bf16
    unsigned int u = __float_as_uint(f);
    unsigned int r = u + 0x7FFFu + ((u >> 16) & 1u);
    return (unsigned short)(r >> 16);
}

__device__ __forceinline__ float neg_inf() { return __uint_as_float(0xFF800000u); }

// A-fragment (16x32, 16-bit) K index for element e of lane l (M = l%16)
__device__ __forceinline__ int kmapA(int lane, int e) {
    return ((e < 8) ? 0 : 16) + ((lane >> 4) << 3) + (e & 7);
}
// B-fragment (32x16, 16-bit) K index for element e of lane l (N = l%16)
__device__ __forceinline__ int kmapB(int lane, int e) {
    return ((e < 8) ? 0 : 16) + ((lane >> 4) << 3) + (e & 7);
}

// ---- W1 pre-pack: (2048x128 f32) -> bf16 B-fragments [kt(64)][nt(8)][lane(32)][e(16)] ----
__global__ void pack_w1(const float* __restrict__ W1, unsigned short* __restrict__ W1p) {
    int idx  = blockIdx.x * blockDim.x + threadIdx.x;   // 262144
    int e    = idx & 15;
    int lane = (idx >> 4) & 31;
    int nt   = (idx >> 9) & 7;
    int kt   = idx >> 12;
    int K    = kt * 32 + kmapB(lane, e);
    int N    = nt * 16 + (lane & 15);
    W1p[idx] = f2bf(W1[(size_t)K * H_DIM + N]);
}

// ---- W2 pre-pack: (128x2048 f32) -> bf16 B-fragments [kt(4)][nt(128)][lane(32)][e(16)] ----
__global__ void pack_w2(const float* __restrict__ W2, unsigned short* __restrict__ W2p) {
    int idx  = blockIdx.x * blockDim.x + threadIdx.x;   // 262144
    int e    = idx & 15;
    int lane = (idx >> 4) & 31;
    int nt   = (idx >> 9) & 127;
    int kt   = idx >> 16;
    int K    = kt * 32 + kmapB(lane, e);
    int N    = nt * 16 + (lane & 15);
    W2p[idx] = f2bf(W2[(size_t)K * C_DIM + N]);
}

// Build per-lane local top-4 (stable: among equal values, smaller local index i —
// i.e. smaller column — stays first) over the 64 register-resident values,
// skipping entries whose bit is set in the removed-mask rm.
__device__ __forceinline__ void scan_top4(const float (&vals)[64], unsigned long long rm,
                                          float& v0, float& v1, float& v2, float& v3,
                                          int& i0, int& i1, int& i2, int& i3)
{
    const float NINF = neg_inf();
    v0 = v1 = v2 = v3 = NINF;
    i0 = i1 = i2 = i3 = -1;
    #pragma unroll
    for (int i = 0; i < 64; ++i) {
        float v = ((rm >> i) & 1ull) ? NINF : vals[i];
        bool b0 = v > v0, b1 = v > v1, b2 = v > v2, b3 = v > v3;
        v3 = b3 ? (b2 ? v2 : v) : v3;  i3 = b3 ? (b2 ? i2 : i) : i3;
        v2 = b2 ? (b1 ? v1 : v) : v2;  i2 = b2 ? (b1 ? i1 : i) : i2;
        v1 = b1 ? (b0 ? v0 : v) : v1;  i1 = b1 ? (b0 ? i0 : i) : i1;
        v0 = b0 ? v : v0;              i0 = b0 ? i : i0;
    }
}

__global__ __launch_bounds__(THREADS)
void se_fused(const float* __restrict__ x,
              const unsigned short* __restrict__ W1p,
              const unsigned short* __restrict__ W2p,
              float* __restrict__ out)
{
    __shared__ float hLDS[ROWS_PER_WG * H_DIM];   // 8 KB
    __shared__ float wS[ROWS_PER_WG * C_DIM];     // 128 KB (CDNA5: up to 320 KB/WG)

    const int lane    = threadIdx.x & 31;
    const int wv      = threadIdx.x >> 5;
    const int rowBase = blockIdx.x * ROWS_PER_WG;
    const int m       = lane & 15;

    for (int i = threadIdx.x; i < ROWS_PER_WG * H_DIM; i += THREADS) hLDS[i] = 0.f;
    __syncthreads();

    // =========== GEMM1: h_partial = x_tile[:, wv*256 : wv*256+256] @ W1[...] ===========
    v8f acc[8];
    #pragma unroll
    for (int nt = 0; nt < 8; ++nt) {
        v8f z = {0.f, 0.f, 0.f, 0.f, 0.f, 0.f, 0.f, 0.f};
        acc[nt] = z;
    }

    const float* xrow = x + (size_t)(rowBase + m) * C_DIM;

    for (int kt = 0; kt < 8; ++kt) {
        const int K0 = wv * 256 + kt * 32 + ((lane >> 4) << 3);
        float4 f0 = *(const float4*)(xrow + K0);
        float4 f1 = *(const float4*)(xrow + K0 + 4);
        float4 f2 = *(const float4*)(xrow + K0 + 16);
        float4 f3 = *(const float4*)(xrow + K0 + 20);
        Frag32 a;
        a.us[0]  = f2bf(f0.x); a.us[1]  = f2bf(f0.y); a.us[2]  = f2bf(f0.z); a.us[3]  = f2bf(f0.w);
        a.us[4]  = f2bf(f1.x); a.us[5]  = f2bf(f1.y); a.us[6]  = f2bf(f1.z); a.us[7]  = f2bf(f1.w);
        a.us[8]  = f2bf(f2.x); a.us[9]  = f2bf(f2.y); a.us[10] = f2bf(f2.z); a.us[11] = f2bf(f2.w);
        a.us[12] = f2bf(f3.x); a.us[13] = f2bf(f3.y); a.us[14] = f2bf(f3.z); a.us[15] = f2bf(f3.w);

        const int ktG = wv * 8 + kt;
        #pragma unroll
        for (int nt = 0; nt < 8; ++nt) {
            Frag32 b;
            const uint4* bp = (const uint4*)(W1p + ((size_t)(ktG * 8 + nt) * 32 + lane) * 16);
            b.q[0] = bp[0];
            b.q[1] = bp[1];
            acc[nt] = __builtin_amdgcn_wmma_f32_16x16x32_bf16(
                false, a.v, false, b.v, (short)0, acc[nt], false, false);
        }
    }

    // reduce K-split partials into hLDS via LDS float atomics (ds_add_f32)
    #pragma unroll
    for (int nt = 0; nt < 8; ++nt) {
        #pragma unroll
        for (int r = 0; r < 8; ++r) {
            int M = r + ((lane >> 4) << 3);
            atomicAdd(&hLDS[M * H_DIM + nt * 16 + (lane & 15)], acc[nt][r]);
        }
    }
    __syncthreads();

    // =========== GEMM2: w = sigmoid(relu(h) @ W2), wave wv owns N-tiles wv*16..wv*16+15 ===========
    v16bf ha[4];
    #pragma unroll
    for (int kt = 0; kt < 4; ++kt) {
        Frag32 a;
        #pragma unroll
        for (int e = 0; e < 16; ++e) {
            int   K = kt * 32 + kmapA(lane, e);
            float v = hLDS[m * H_DIM + K];
            a.us[e] = f2bf(v > 0.f ? v : 0.f);
        }
        ha[kt] = a.v;
    }

    for (int t = 0; t < 16; ++t) {
        const int nt = wv * 16 + t;
        if (t < 15) {   // warm L0 for next tile's B-fragments (global_prefetch_b8)
            __builtin_prefetch(W2p + ((size_t)(nt + 1) * 32 + lane) * 16, 0, 3);
        }
        v8f c = {0.f, 0.f, 0.f, 0.f, 0.f, 0.f, 0.f, 0.f};
        #pragma unroll
        for (int kt = 0; kt < 4; ++kt) {
            Frag32 b;
            const uint4* bp = (const uint4*)(W2p + ((size_t)(kt * 128 + nt) * 32 + lane) * 16);
            b.q[0] = bp[0];
            b.q[1] = bp[1];
            c = __builtin_amdgcn_wmma_f32_16x16x32_bf16(
                false, ha[kt], false, b.v, (short)0, c, false, false);
        }
        #pragma unroll
        for (int r = 0; r < 8; ++r) {
            float s = 1.f / (1.f + __expf(-c[r]));
            int   M = r + ((lane >> 4) << 3);
            wS[M * C_DIM + nt * 16 + (lane & 15)] = s;
        }
    }
    __syncthreads();

    // =========== selection: wave wv handles rows wv and wv+8 ===========
    for (int rr = 0; rr < 2; ++rr) {
        const int row = wv + rr * 8;

        float vals[64];                     // lane holds columns lane, lane+32, ...
        #pragma unroll
        for (int i = 0; i < 64; ++i) vals[i] = wS[row * C_DIM + i * 32 + lane];
        const float myw0 = vals[0];         // w[row, lane] for lane < 32

        // ---- exact stable ranks of columns 0..31 (lane c gets rank of column c) ----
        // Ballot/popcount form: the compare is 1 VALU (v_cmp -> SGPR); the popcount
        // and accumulate run on the SALU, co-issuing with the VALU stream, and the
        // result is already wave-uniform (no shuffle reduction needed).
        // Tie-break (v==t && col<c) can only fire for i==0 (col=lane<32): one masked ballot.
        int myrank = 0;
        for (int c = 0; c < 32; ++c) {
            float t = __shfl(myw0, c, 32);
            unsigned long long eq = __ballot(vals[0] == t);
            int cnt = (int)__popcll(eq & ((1ull << c) - 1ull));
            #pragma unroll
            for (int i = 0; i < 64; ++i) {
                cnt += (int)__popcll(__ballot(vals[i] > t));
            }
            myrank = (lane == c) ? cnt : myrank;
        }

        // ---- top-32 values descending (stable): lane i gets w_top[i] ----
        // Per-lane sorted top-4 buffer + wave argmax tournament; masked rescan only
        // when a lane's buffer empties while it still has live values.
        float wtop = 0.f;
        unsigned long long rm = 0ull;       // consumed local slots
        int consumed = 0;
        float v0, v1, v2, v3;
        int   i0, i1, i2, i3;
        scan_top4(vals, rm, v0, v1, v2, v3, i0, i1, i2, i3);

        for (int it = 0; it < 32; ++it) {
            bool need = (i0 < 0) && (consumed < 64);
            if (__ballot(need)) {
                if (need) scan_top4(vals, rm, v0, v1, v2, v3, i0, i1, i2, i3);
            }
            float bv = v0;
            int   bc = (i0 < 0) ? 0x7FFFFFFF : (i0 * 32 + lane);
            #pragma unroll
            for (int off = 16; off > 0; off >>= 1) {
                float vv   = __shfl_xor(bv, off, 32);
                int   cc   = __shfl_xor(bc, off, 32);
                bool  take = (vv > bv) || (vv == bv && cc < bc);
                bv = take ? vv : bv;
                bc = take ? cc : bc;
            }
            if (lane == it) wtop = bv;
            if ((bc & 31) == lane) {        // this lane owns the extracted element
                rm |= 1ull << (bc >> 5);
                ++consumed;
                v0 = v1; i0 = i1;
                v1 = v2; i1 = i2;
                v2 = v3; i2 = i3;
                v3 = neg_inf(); i3 = -1;
            }
        }

        // ---- cols = ascending sort of the 32 (distinct) ranks, via shuffles only ----
        int pos = 0;
        #pragma unroll
        for (int s = 0; s < 32; ++s) {
            int r2 = __shfl(myrank, s, 32);
            pos += (r2 < myrank) ? 1 : 0;
        }
        int mycol = 0;
        #pragma unroll
        for (int s = 0; s < 32; ++s) {
            int r2 = __shfl(myrank, s, 32);
            int p2 = __shfl(pos, s, 32);
            mycol = (p2 == lane) ? r2 : mycol;
        }

        // ---- out[row, i] = x[row, cols[i]] * w_top[i] ----
        float xv = x[(size_t)(rowBase + row) * C_DIM + mycol];
        out[(size_t)(rowBase + row) * K_TOP + lane] = xv * wtop;
    }
}

extern "C" void kernel_launch(void* const* d_in, const int* in_sizes, int n_in,
                              void* d_out, int out_size, void* d_ws, size_t ws_size,
                              hipStream_t stream) {
    const float* x  = (const float*)d_in[0];
    const float* W1 = (const float*)d_in[1];
    const float* W2 = (const float*)d_in[2];
    (void)in_sizes; (void)n_in; (void)out_size; (void)ws_size;

    unsigned short* W1p = (unsigned short*)d_ws;            // 512 KB
    unsigned short* W2p = W1p + 262144;                     // 512 KB

    pack_w1<<<262144 / 256, 256, 0, stream>>>(W1, W1p);
    pack_w2<<<262144 / 256, 256, 0, stream>>>(W2, W2p);
    se_fused<<<N_ROWS / ROWS_PER_WG, THREADS, 0, stream>>>(x, W1p, W2p, (float*)d_out);
}
